// MultiHeadSelfAttention_47614007443924
// MI455X (gfx1250) — compile-verified, matrix operand order checked by reading, not run on hardware
//
#include <hip/hip_runtime.h>
#include <stdint.h>

#define BB 4
#define NN 512
#define DD 64
#define HH 8

typedef __attribute__((ext_vector_type(16))) __bf16 v16bf;
typedef __attribute__((ext_vector_type(8)))  float  v8f;
typedef __attribute__((ext_vector_type(4)))  unsigned int u32x4;
typedef __attribute__((ext_vector_type(8)))  unsigned int u32x8;
typedef __attribute__((ext_vector_type(8)))  int i32x8;
typedef __attribute__((ext_vector_type(4)))  int i32x4;

// Pack two f32 into one dword of two bf16 (truncating round): elem j even = low half.
__device__ __forceinline__ unsigned pk2bf(float lo, float hi) {
  const unsigned a = __builtin_bit_cast(unsigned, lo);
  const unsigned b = __builtin_bit_cast(unsigned, hi);
  return (a >> 16) | (b & 0xffff0000u);
}

// v_wmma_*_16x16x32 16-bit operand K-layout (ISA 7.12.2):
// lanes 0-15: elems 0..7 -> K=kb+0..7,  elems 8..15 -> K=kb+16..23
// lanes16-31: elems 0..7 -> K=kb+8..15, elems 8..15 -> K=kb+24..31
// => per lane: two runs of 8 consecutive K values.
__device__ __forceinline__ v16bf bf16_from_2x8(const float* lo8, const float* hi8) {
  const float4 a = *(const float4*)lo8;
  const float4 b = *(const float4*)(lo8 + 4);
  const float4 c = *(const float4*)hi8;
  const float4 d = *(const float4*)(hi8 + 4);
  u32x8 u = { pk2bf(a.x, a.y), pk2bf(a.z, a.w), pk2bf(b.x, b.y), pk2bf(b.z, b.w),
              pk2bf(c.x, c.y), pk2bf(c.z, c.w), pk2bf(d.x, d.y), pk2bf(d.z, d.w) };
  return __builtin_bit_cast(v16bf, u);
}

// Same but source elements strided (for weight operands).
__device__ __forceinline__ v16bf bf16_strided(const float* p, int stride,
                                              int k_lo, int k_hi) {
  u32x8 u = {
    pk2bf(p[(k_lo + 0) * stride], p[(k_lo + 1) * stride]),
    pk2bf(p[(k_lo + 2) * stride], p[(k_lo + 3) * stride]),
    pk2bf(p[(k_lo + 4) * stride], p[(k_lo + 5) * stride]),
    pk2bf(p[(k_lo + 6) * stride], p[(k_lo + 7) * stride]),
    pk2bf(p[(k_hi + 0) * stride], p[(k_hi + 1) * stride]),
    pk2bf(p[(k_hi + 2) * stride], p[(k_hi + 3) * stride]),
    pk2bf(p[(k_hi + 4) * stride], p[(k_hi + 5) * stride]),
    pk2bf(p[(k_hi + 6) * stride], p[(k_hi + 7) * stride]) };
  return __builtin_bit_cast(v16bf, u);
}

__device__ __forceinline__ v8f wmma_bf16(v16bf a, v16bf b, v8f c) {
  return __builtin_amdgcn_wmma_f32_16x16x32_bf16(
      false, a, false, b, (short)0, c, false, false);
}

// ---------------------------------------------------------------------------
// Kernel A: QKV = n[2048,64] x {W_q|W_k|W_v}[64,64]  (bf16 WMMA, fp32 acc)
// grid = 128 row-tiles, block = 384 (12 waves: 3 matrices x 4 col-tiles)
// ---------------------------------------------------------------------------
__global__ void qkv_kernel(const float* __restrict__ nmat,
                           const float* __restrict__ Wq,
                           const float* __restrict__ Wk,
                           const float* __restrict__ Wv,
                           float* __restrict__ qkv) {
  const int lane  = threadIdx.x & 31;
  const int wv    = threadIdx.x >> 5;   // 0..11
  const int which = wv >> 2;            // 0=Q,1=K,2=V
  const int cb    = (wv & 3) * 16;      // column base in 64
  const int col   = lane & 15;
  const int off8  = (lane >= 16) ? 8 : 0;

  const float* W = (which == 0) ? Wq : (which == 1) ? Wk : Wv;
  float* outp = qkv + (size_t)which * (BB * NN * DD);

  const float* nr = nmat + (size_t)(blockIdx.x * 16 + col) * DD;
  const v16bf a0 = bf16_from_2x8(nr + off8,      nr + 16 + off8);
  const v16bf a1 = bf16_from_2x8(nr + 32 + off8, nr + 48 + off8);
  const v16bf b0 = bf16_strided(W + cb + col, DD, off8,      16 + off8);
  const v16bf b1 = bf16_strided(W + cb + col, DD, 32 + off8, 48 + off8);

  v8f c; for (int i = 0; i < 8; ++i) c[i] = 0.f;
  c = wmma_bf16(a0, b0, c);
  c = wmma_bf16(a1, b1, c);

  const int rbase = blockIdx.x * 16 + off8;
  for (int r = 0; r < 8; ++r)
    outp[(size_t)(rbase + r) * DD + cb + col] = c[r];
}

// ---------------------------------------------------------------------------
// Kernel B: per (b,n) workgroup (2048 WGs x 256 thr):
//   TDM-load e[b,n,:,:] (512x64 f32, 128 KB) into LDS; per 16-row m-tile:
//   [E|G] = e_tile x [W_e|W_g] (2 WMMAs), sigmoid + ds_add_f32 centrality,
//   Eb = clip(QK/sqrt(dk)) + E kept in LDS, e_out = Eb x O_e (4 WMMAs).
//   Then per-head softmax, P@V, x log1p(sum G), n_out = Vh @ O_n.
// ---------------------------------------------------------------------------
__global__ void edge_attn_kernel(const float* __restrict__ e,
                                 const float* __restrict__ qkv,
                                 const float* __restrict__ W_e,
                                 const float* __restrict__ W_g,
                                 const float* __restrict__ O_e,
                                 const float* __restrict__ O_n,
                                 float* __restrict__ n_out,
                                 float* __restrict__ e_out) {
  extern __shared__ float smem[];
  float* lds_e  = smem;                 // 512*64 = 32768 f (128 KB)
  float* lds_Eb = smem + NN * DD;       // 8*512  =  4096 f (16 KB)
  float* gsum   = lds_Eb + HH * NN;     // 8 f
  float* vh     = gsum + HH;            // 64 f

  const int bn   = blockIdx.x;          // b*512 + n
  const int b    = bn >> 9;
  const int lane = threadIdx.x & 31;
  const int wave = threadIdx.x >> 5;
  const int col  = lane & 15;
  const int off8 = (lane >= 16) ? 8 : 0;

  // ---- TDM: async DMA of the e slab into LDS (wave 0 issues; wait deferred) --
  if (wave == 0) {
    if (lane < HH) gsum[lane] = 0.f;
    const unsigned long long ga =
        (unsigned long long)(uintptr_t)e + (unsigned long long)bn * (NN * DD * 4ull);
    u32x4 g0;
    g0.x = 1u;                                      // count=1 (valid user D#)
    g0.y = 0u;                                      // lds_addr = dyn-LDS base
    g0.z = (unsigned)(ga & 0xffffffffu);            // global_addr[31:0]
    g0.w = (unsigned)((ga >> 32) & 0x01ffffffu) | (2u << 30); // [56:32] | type=2
    i32x8 g1;
    g1[0] = (int)(2u << 16);   // data_size = 4B
    g1[1] = (int)(DD << 16);   // tensor_dim0 = 64   (bits 79:48, low half)
    g1[2] = (int)(NN << 16);   // tensor_dim1 = 512  (bits 111:80, low half)
    g1[3] = (int)(DD << 16);   // tile_dim0 = 64     (bits 127:112)
    g1[4] = NN;                // tile_dim1 = 512    (bits 143:128)
    g1[5] = DD;                // tensor_dim0_stride = 64 (bits 207:160)
    g1[6] = 0;
    g1[7] = 0;
    i32x4 gz4; gz4[0] = 0; gz4[1] = 0; gz4[2] = 0; gz4[3] = 0;
    i32x8 gz8; for (int i = 0; i < 8; ++i) gz8[i] = 0;
    __builtin_amdgcn_tensor_load_to_lds(g0, g1, gz4, gz4, gz8, 0);
  }

  // ---- per-wave invariant operands (overlap with TDM in flight) ----
  // [W_e | W_g] packed as 64x16 (cols 0..7 = E, 8..15 = G), K-chunks of 32
  const float* wcol = (col < 8) ? (W_e + col) : (W_g + (col - 8));
  const v16bf beg0 = bf16_strided(wcol, HH, off8,      16 + off8);
  const v16bf beg1 = bf16_strided(wcol, HH, 32 + off8, 48 + off8);

  // O_e [8x64] zero-padded to K=32, split into 4 N-chunks of 16:
  // only lanes 0-15, elems 0..7 carry data (K=h=0..7); rest are padding zeros.
  v16bf boe[4];
  for (int ch = 0; ch < 4; ++ch) {
    const float* p = O_e + ch * 16 + col;   // stride DD over h
    const bool lo = (lane < 16);
    u32x8 u = { lo ? pk2bf(p[0 * DD], p[1 * DD]) : 0u,
                lo ? pk2bf(p[2 * DD], p[3 * DD]) : 0u,
                lo ? pk2bf(p[4 * DD], p[5 * DD]) : 0u,
                lo ? pk2bf(p[6 * DD], p[7 * DD]) : 0u,
                0u, 0u, 0u, 0u };
    boe[ch] = __builtin_bit_cast(v16bf, u);
  }

  const float* q_ws = qkv;
  const float* k_ws = qkv + (size_t)BB * NN * DD;
  const float* v_ws = qkv + (size_t)2 * BB * NN * DD;
  float4 qv0 = {0.f, 0.f, 0.f, 0.f}, qv1 = {0.f, 0.f, 0.f, 0.f};
  if (col < 8) {
    const float* qp = q_ws + (size_t)bn * DD + col * 8;
    qv0 = *(const float4*)qp;
    qv1 = *(const float4*)(qp + 4);
  }

  if (wave == 0) __builtin_amdgcn_s_wait_tensorcnt(0);
  __syncthreads();  // e slab resident in LDS

  // ---- stage 1: stream m-tiles ----
  for (int t = wave; t < NN / 16; t += 8) {
    const int m0 = t * 16;

    // A operand: e_tile [16m x 64d] fp32->bf16 straight from the LDS slab
    const float* er = lds_e + (size_t)(m0 + col) * DD;
    const v16bf a0 = bf16_from_2x8(er + off8,      er + 16 + off8);
    const v16bf a1 = bf16_from_2x8(er + 32 + off8, er + 48 + off8);
    v8f c; for (int i = 0; i < 8; ++i) c[i] = 0.f;
    c = wmma_bf16(a0, beg0, c);
    c = wmma_bf16(a1, beg1, c);

    const int mbase = m0 + off8;
    if (col >= 8) {             // gate columns: sigmoid + centrality sum
      float s = 0.f;
      for (int r = 0; r < 8; ++r) s += 1.f / (1.f + __expf(-c[r]));
      atomicAdd(&gsum[col - 8], s);    // ds_add_f32
    } else {                    // E columns: clipped scores + edge bias -> Eb
      const int h = col;
      for (int r = 0; r < 8; ++r) {
        const int m = mbase + r;
        const float* kr = k_ws + (size_t)(b * NN + m) * DD + h * 8;
        const float4 k0 = *(const float4*)kr;
        const float4 k1 = *(const float4*)(kr + 4);
        float dot = qv0.x*k0.x + qv0.y*k0.y + qv0.z*k0.z + qv0.w*k0.w
                  + qv1.x*k1.x + qv1.y*k1.y + qv1.z*k1.z + qv1.w*k1.w;
        dot *= 0.35355339059327373f;             // dk^-0.5
        dot = fminf(5.f, fmaxf(-5.f, dot));
        lds_Eb[h * NN + m] = dot + c[r];
      }
    }

    // e_out tile [16m x 64d] = Eb[16x8 pad 32] x O_e, 4 WMMAs
    {
      const float* p = lds_Eb + m0 + col;   // stride NN over h
      const bool lo = (lane < 16);
      u32x8 u = { lo ? pk2bf(p[0 * NN], p[1 * NN]) : 0u,
                  lo ? pk2bf(p[2 * NN], p[3 * NN]) : 0u,
                  lo ? pk2bf(p[4 * NN], p[5 * NN]) : 0u,
                  lo ? pk2bf(p[6 * NN], p[7 * NN]) : 0u,
                  0u, 0u, 0u, 0u };
      const v16bf ap = __builtin_bit_cast(v16bf, u);
      float* eo = e_out + (size_t)bn * (NN * DD);
      for (int ch = 0; ch < 4; ++ch) {
        v8f cc; for (int i = 0; i < 8; ++i) cc[i] = 0.f;
        cc = wmma_bf16(ap, boe[ch], cc);
        const int d = ch * 16 + col;
        for (int r = 0; r < 8; ++r)
          eo[(size_t)(mbase + r) * DD + d] = cc[r];
      }
    }
  }
  __syncthreads();

  // ---- stage 2: one wave per head: softmax over m, P@V, centrality scale ----
  {
    const int h = wave;
    float mx = -1e30f;
    for (int m = lane; m < NN; m += 32) mx = fmaxf(mx, lds_Eb[h * NN + m]);
    for (int off = 16; off > 0; off >>= 1) mx = fmaxf(mx, __shfl_xor(mx, off, 32));
    float sum = 0.f;
    for (int m = lane; m < NN; m += 32) {
      const float p = __expf(lds_Eb[h * NN + m] - mx);
      lds_Eb[h * NN + m] = p;          // wave-exclusive row
      sum += p;
    }
    for (int off = 16; off > 0; off >>= 1) sum += __shfl_xor(sum, off, 32);
    const float inv = 1.f / sum;
    float acc[8];
    for (int j = 0; j < 8; ++j) acc[j] = 0.f;
    for (int m = lane; m < NN; m += 32) {
      const float p = lds_Eb[h * NN + m] * inv;
      const float* vr = v_ws + (size_t)(b * NN + m) * DD + h * 8;
      const float4 v0 = *(const float4*)vr;
      const float4 v1 = *(const float4*)(vr + 4);
      acc[0] += p * v0.x; acc[1] += p * v0.y; acc[2] += p * v0.z; acc[3] += p * v0.w;
      acc[4] += p * v1.x; acc[5] += p * v1.y; acc[6] += p * v1.z; acc[7] += p * v1.w;
    }
    for (int off = 16; off > 0; off >>= 1)
      for (int j = 0; j < 8; ++j) acc[j] += __shfl_xor(acc[j], off, 32);
    if (lane == 0) {
      const float cent = __logf(1.f + gsum[h]);   // log1p(sum_m G)
      for (int j = 0; j < 8; ++j) vh[h * 8 + j] = acc[j] * cent;
    }
  }
  __syncthreads();

  // ---- stage 3: n_out[b,n,:] = vh[64] @ O_n[64,64] ----
  if (wave == 0) {
    for (int cidx = lane; cidx < DD; cidx += 32) {
      float s = 0.f;
      for (int k = 0; k < DD; ++k) s += vh[k] * O_n[k * DD + cidx];
      n_out[(size_t)bn * DD + cidx] = s;
    }
  }
}

extern "C" void kernel_launch(void* const* d_in, const int* in_sizes, int n_in,
                              void* d_out, int out_size, void* d_ws, size_t ws_size,
                              hipStream_t stream) {
  const float* nmat_in = (const float*)d_in[0];
  const float* e_in    = (const float*)d_in[1];
  const float* W_q     = (const float*)d_in[2];
  const float* W_k     = (const float*)d_in[3];
  const float* W_v     = (const float*)d_in[4];
  const float* O_n     = (const float*)d_in[5];
  const float* W_e     = (const float*)d_in[6];
  const float* W_g     = (const float*)d_in[7];
  const float* O_e     = (const float*)d_in[8];

  float* out   = (float*)d_out;
  float* n_out = out;                         // [4,512,64]
  float* e_out = out + (size_t)BB * NN * DD;  // [4,512,512,64]
  float* qkv   = (float*)d_ws;                // 3 * 4*512*64 f = 1.5 MB

  qkv_kernel<<<(BB * NN) / 16, 384, 0, stream>>>(nmat_in, W_q, W_k, W_v, qkv);

  const size_t smem = (size_t)(NN * DD + HH * NN + HH + 64) * sizeof(float); // ~148 KB
  edge_attn_kernel<<<BB * NN, 256, smem, stream>>>(e_in, qkv, W_e, W_g, O_e,
                                                   O_n, n_out, e_out);
}
